// selfAttention_49435073577628
// MI455X (gfx1250) — compile-verified, matrix operand order checked by reading, not run on hardware
//
#include <hip/hip_runtime.h>
#include <hip/hip_bf16.h>

typedef __attribute__((ext_vector_type(16))) __bf16 v16bf;
typedef __attribute__((ext_vector_type(8)))  __bf16 v8bf;
typedef __attribute__((ext_vector_type(4)))  __bf16 v4bf;
typedef __attribute__((ext_vector_type(2)))  __bf16 v2bf;
typedef __attribute__((ext_vector_type(8)))  float  v8f;

#define NH      16
#define HD      64
#define S_LEN   2048
#define DMODEL  1024
#define BATCH   2
#define MROWS   (BATCH * S_LEN)   // 4096

// softmax done base-2: scale = (1/sqrt(64)) * log2(e), folded into Q at the
// QKV-projection epilogue (Q is rounded to bf16 there anyway, so it's free).
#define Q_SCALE 0.18033688011112042f
#define NEG_BIG (-1.0e30f)

// ---------------------------------------------------------------------------
// f32 -> bf16 conversion (vectorized x4). n must be a multiple of 4.
// ---------------------------------------------------------------------------
__global__ void f32_to_bf16_kernel(const float* __restrict__ in,
                                   __bf16* __restrict__ out, int n) {
  int i = (blockIdx.x * blockDim.x + threadIdx.x) * 4;
  if (i + 3 < n) {
    float4 v = *(const float4*)(in + i);
    v4bf o;
    o[0] = (__bf16)v.x; o[1] = (__bf16)v.y;
    o[2] = (__bf16)v.z; o[3] = (__bf16)v.w;
    *(v4bf*)(out + i) = o;
  }
}

// ---------------------------------------------------------------------------
// Fragment loader for V_WMMA_F32_16X16X32_BF16.
// A-frag lane layout: row = rowbase + (lane&15); halves 0..7 hold
// K = k0 + (lane>>4)*8 + {0..7}; halves 8..15 hold K = +16 more.
// B-frag (for C = X @ W^T) has identical layout over rows of W.
// Both are two contiguous 16-byte loads from a row-major [*, ldk] matrix.
// ---------------------------------------------------------------------------
__device__ __forceinline__ v16bf load_frag(const __bf16* __restrict__ base,
                                           int rowbase, int ldk, int k0,
                                           int lane) {
  const __bf16* p = base + (size_t)(rowbase + (lane & 15)) * ldk
                         + k0 + ((lane >> 4) << 3);
  v16bf f;
  *((v8bf*)&f)       = *(const v8bf*)(p);
  *(((v8bf*)&f) + 1) = *(const v8bf*)(p + 16);
  return f;
}

__device__ __forceinline__ v16bf load_frag_lds(const __bf16* p) {
  v16bf f;
  *((v8bf*)&f)       = *(const v8bf*)(p);
  *(((v8bf*)&f) + 1) = *(const v8bf*)(p + 16);
  return f;
}

__device__ __forceinline__ v8f wmma_bf16(v16bf a, v16bf b, v8f c) {
  return __builtin_amdgcn_wmma_f32_16x16x32_bf16(false, a, false, b,
                                                 (short)0, c, false, false);
}

// Max-reduce across the 16-lane half-groups with DPP ROW_XMASK (VALU pipe,
// no LDS traffic). Lanes 0-15 and 16-31 reduce independently, which matches
// the WMMA C-layout row groups.
__device__ __forceinline__ float rowmax16(float x) {
  float t;
  t = __int_as_float(__builtin_amdgcn_update_dpp(
      0, __float_as_int(x), 0x161, 0xf, 0xf, true));   // xor 1
  x = fmaxf(x, t);
  t = __int_as_float(__builtin_amdgcn_update_dpp(
      0, __float_as_int(x), 0x162, 0xf, 0xf, true));   // xor 2
  x = fmaxf(x, t);
  t = __int_as_float(__builtin_amdgcn_update_dpp(
      0, __float_as_int(x), 0x164, 0xf, 0xf, true));   // xor 4
  x = fmaxf(x, t);
  t = __int_as_float(__builtin_amdgcn_update_dpp(
      0, __float_as_int(x), 0x168, 0xf, 0xf, true));   // xor 8
  x = fmaxf(x, t);
  return x;
}

// ---------------------------------------------------------------------------
// GEMM:  C[M,N] = A[M,K](bf16) @ W[N,K]^T(bf16) + bias[N](f32)
// Block: 256 threads = 8 waves, block tile 128(M) x 256(N).
// Wave tile: 64 x 64 = 4x4 WMMA accumulators -> 16 WMMAs per 8 fragment
// loads per k-step.
// QKV=true : scatter into head-major bf16 QKV buffer [part][b][h][s][hd]
//            (part stride MROWS*DMODEL); Q additionally scaled by Q_SCALE.
// QKV=false: plain f32 [M,N] output (final projection).
// ---------------------------------------------------------------------------
template <bool QKV>
__global__ __launch_bounds__(256, 1) void gemm_bf16_kernel(
    const __bf16* __restrict__ A, const __bf16* __restrict__ W,
    const float* __restrict__ bias, void* __restrict__ out,
    int M, int N, int K) {
  const int lane = threadIdx.x & 31;
  const int wave = threadIdx.x >> 5;
  const int wm = wave & 1;   // M band (64 rows)
  const int wn = wave >> 1;  // N band (64 cols)
  const int m0 = blockIdx.y * 128 + wm * 64;
  const int n0 = blockIdx.x * 256 + wn * 64;

  v8f acc[4][4];
#pragma unroll
  for (int i = 0; i < 4; ++i)
#pragma unroll
    for (int j = 0; j < 4; ++j)
#pragma unroll
      for (int r = 0; r < 8; ++r) acc[i][j][r] = 0.0f;

  for (int k0 = 0; k0 < K; k0 += 32) {
    v16bf a[4], b[4];
#pragma unroll
    for (int i = 0; i < 4; ++i) a[i] = load_frag(A, m0 + i * 16, K, k0, lane);
#pragma unroll
    for (int j = 0; j < 4; ++j) b[j] = load_frag(W, n0 + j * 16, K, k0, lane);
#pragma unroll
    for (int i = 0; i < 4; ++i)
#pragma unroll
      for (int j = 0; j < 4; ++j)
        acc[i][j] = wmma_bf16(a[i], b[j], acc[i][j]);
  }

  // C layout: vgpr r, lanes 0-15 -> M=r, lanes 16-31 -> M=r+8; N = lane&15.
  const int mh = (lane >> 4) << 3;
  const int nn = lane & 15;
#pragma unroll
  for (int i = 0; i < 4; ++i) {
#pragma unroll
    for (int j = 0; j < 4; ++j) {
      const int n = n0 + j * 16 + nn;
      const float bv = bias[n];
#pragma unroll
      for (int r = 0; r < 8; ++r) {
        const int m = m0 + i * 16 + mh + r;
        float v = acc[i][j][r] + bv;
        if (QKV) {
          const int part = n >> 10;          // 0=Q 1=K 2=V
          if (part == 0) v *= Q_SCALE;       // fold softmax scale into Q
          const int d = n & 1023;
          const int h = d >> 6, hd = d & 63;
          const int bb = m >> 11, s = m & 2047;
          __bf16* dst = (__bf16*)out;
          dst[(size_t)part * ((size_t)MROWS * DMODEL)
              + ((size_t)(bb * NH + h) * S_LEN + s) * HD + hd] = (__bf16)v;
        } else {
          ((float*)out)[(size_t)m * N + n] = v;
        }
      }
    }
  }
}

// ---------------------------------------------------------------------------
// Causal flash attention over head-major bf16 Q/K/V (Q pre-scaled, base-2
// softmax). Grid: (S/128, B*H). Block: 256 threads = 8 waves; each wave owns
// 16 query rows (block covers 128). Key loop in 64-key tiles shared by all
// waves (one V staging + barrier pair per 64 keys for 128 query rows).
// Per active wave per tile: 8 WMMAs QK^T, DPP row-max, exp2, P->LDS->A-frag,
// 2 WMMAs row-sum (P @ ones), 8 WMMAs P@V against V staged transposed in LDS.
// Causal mask only where the tile straddles the wave's diagonal.
// ---------------------------------------------------------------------------
__global__ __launch_bounds__(256) void attn_kernel(
    const __bf16* __restrict__ QKV, __bf16* __restrict__ Oattn) {
  const int lane = threadIdx.x & 31;
  const int wave = threadIdx.x >> 5;         // 0..7
  const int bh = blockIdx.y;                 // 0..31
  const int b = bh >> 4, h = bh & 15;
  const int qbase = blockIdx.x * 128;
  const int qw = qbase + wave * 16;          // this wave's first query row

  const size_t headStride = (size_t)S_LEN * HD;
  const size_t partStride = (size_t)MROWS * DMODEL;
  const __bf16* Qh = QKV + (size_t)bh * headStride;
  const __bf16* Kh = QKV + partStride + (size_t)bh * headStride;
  const __bf16* Vh = QKV + 2 * partStride + (size_t)bh * headStride;

  // Row stride 72 halves = 144 bytes (16B-aligned rows for b128 ds loads,
  // conflict-free across the 16-lane fragment groups).
  __shared__ __attribute__((aligned(16))) __bf16 Vt[64][72];     // [hd][key]
  __shared__ __attribute__((aligned(16))) __bf16 Plds[8][16][72];// [q][key]

  const v16bf q0 = load_frag(Qh, qw, HD, 0, lane);
  const v16bf q1 = load_frag(Qh, qw, HD, 32, lane);

  v16bf ones;
#pragma unroll
  for (int e = 0; e < 16; ++e) ones[e] = (__bf16)1.0f;

  float mstat[8], lstat[8];
  v8f oacc[4];
#pragma unroll
  for (int r = 0; r < 8; ++r) { mstat[r] = NEG_BIG; lstat[r] = 0.0f; }
#pragma unroll
  for (int c = 0; c < 4; ++c)
#pragma unroll
    for (int r = 0; r < 8; ++r) oacc[c][r] = 0.0f;

  const int mh = (lane >> 4) << 3;
  const int nn = lane & 15;
  const int ntiles = 2 * blockIdx.x + 2;     // keys 0 .. qbase+127

  for (int t = 0; t < ntiles; ++t) {
    const int n0 = t << 6;

    // ---- stage V[n0..n0+63][0..63] transposed into LDS ----
    // Each thread handles one key-pair x 8 hd values: 2x b128 global loads,
    // 8x packed b32 LDS stores (two keys per dword).
    __syncthreads();                         // protect prior Vt readers
    {
      const int kp = threadIdx.x >> 3;               // key pair 0..31
      const int hd0 = (threadIdx.x & 7) << 3;        // 0..56
      const __bf16* vsrc = Vh + (size_t)(n0 + 2 * kp) * HD + hd0;
      const v8bf cA = *(const v8bf*)(vsrc);
      const v8bf cB = *(const v8bf*)(vsrc + HD);
#pragma unroll
      for (int e = 0; e < 8; ++e) {
        v2bf pr; pr[0] = cA[e]; pr[1] = cB[e];
        *(v2bf*)&Vt[hd0 + e][2 * kp] = pr;
      }
    }
    __syncthreads();

    if (n0 <= qw + 15) {                     // wave-uniform: EXEC stays full
      // ---- scores: 4 sub-tiles of 16 keys, Hd=64 in 2 k-steps each ----
      v8f s[4];
#pragma unroll
      for (int j = 0; j < 4; ++j) {
        const v16bf kb0 = load_frag(Kh, n0 + j * 16, HD, 0, lane);
        const v16bf kb1 = load_frag(Kh, n0 + j * 16, HD, 32, lane);
        v8f z;
#pragma unroll
        for (int r = 0; r < 8; ++r) z[r] = 0.0f;
        s[j] = wmma_bf16(q1, kb1, wmma_bf16(q0, kb0, z));
      }

      // ---- causal mask: only where the tile straddles this wave's rows ----
      if (n0 + 63 > qw) {
#pragma unroll
        for (int j = 0; j < 4; ++j)
#pragma unroll
          for (int r = 0; r < 8; ++r) {
            const int qg = qw + mh + r;
            if (n0 + j * 16 + nn > qg) s[j][r] = NEG_BIG;
          }
      }

      // ---- online softmax: DPP row-max, base-2 exp ----
      float alpha[8];
#pragma unroll
      for (int r = 0; r < 8; ++r) {
        float rv = fmaxf(fmaxf(s[0][r], s[1][r]), fmaxf(s[2][r], s[3][r]));
        rv = rowmax16(rv);
        const float mnew = fmaxf(mstat[r], rv);
        alpha[r] = __builtin_amdgcn_exp2f(mstat[r] - mnew);
        mstat[r] = mnew;
#pragma unroll
        for (int j = 0; j < 4; ++j)
          s[j][r] = __builtin_amdgcn_exp2f(s[j][r] - mnew);
      }

      // ---- stage P (bf16) into per-wave LDS, C-layout -> row-major ----
#pragma unroll
      for (int j = 0; j < 4; ++j)
#pragma unroll
        for (int r = 0; r < 8; ++r)
          Plds[wave][mh + r][j * 16 + nn] = (__bf16)s[j][r];
      asm volatile("s_wait_dscnt 0" ::: "memory");   // wave-local LDS ordering

      // ---- re-load P as two A-fragments (16x32 each, k = keys) ----
      const __bf16* pp = &Plds[wave][lane & 15][(lane >> 4) << 3];
      const v16bf pa0 = load_frag_lds(pp);
      const v16bf pa1 = load_frag_lds(pp + 32);

      // ---- row sums via WMMA: P @ ones lands in stats layout ----
      {
        v8f z;
#pragma unroll
        for (int r = 0; r < 8; ++r) z[r] = 0.0f;
        const v8f rs = wmma_bf16(pa1, ones, wmma_bf16(pa0, ones, z));
#pragma unroll
        for (int r = 0; r < 8; ++r)
          lstat[r] = lstat[r] * alpha[r] + rs[r];
      }

      // ---- rescale O and accumulate P @ V (4 Hd chunks of 16) ----
#pragma unroll
      for (int c = 0; c < 4; ++c) {
        v8f a = oacc[c];
#pragma unroll
        for (int r = 0; r < 8; ++r) a[r] *= alpha[r];
        const __bf16* vp = &Vt[c * 16 + (lane & 15)][(lane >> 4) << 3];
        const v16bf vb0 = load_frag_lds(vp);
        const v16bf vb1 = load_frag_lds(vp + 32);
        oacc[c] = wmma_bf16(pa1, vb1, wmma_bf16(pa0, vb0, a));
      }
    }
  }

  // ---- epilogue: O * (1/l) -> bf16 [B,S,D] with D index = h*64 + hd ----
  float rinv[8];
#pragma unroll
  for (int r = 0; r < 8; ++r) rinv[r] = __builtin_amdgcn_rcpf(lstat[r]);
#pragma unroll
  for (int c = 0; c < 4; ++c) {
#pragma unroll
    for (int r = 0; r < 8; ++r) {
      const int s = qw + mh + r;
      const int hd = c * 16 + nn;
      const float v = oacc[c][r] * rinv[r];
      Oattn[((size_t)(b * S_LEN + s)) * DMODEL + h * HD + hd] = (__bf16)v;
    }
  }
}

// ---------------------------------------------------------------------------
// Launch. Workspace layout (bf16 elements), total 48 MB:
//   Xb     : MROWS*DMODEL            (4M)
//   Wqkvb  : 3*DMODEL*DMODEL         (3M)
//   Woutb  : DMODEL*DMODEL           (1M)
//   QKVh   : 3*MROWS*DMODEL          (12M)  head-major [part][b][h][s][hd]
//   Attn   : MROWS*DMODEL            (4M)   [B,S,D]
// ---------------------------------------------------------------------------
extern "C" void kernel_launch(void* const* d_in, const int* in_sizes, int n_in,
                              void* d_out, int out_size, void* d_ws,
                              size_t ws_size, hipStream_t stream) {
  const float* x    = (const float*)d_in[0];
  const float* Wqkv = (const float*)d_in[1];
  const float* bqkv = (const float*)d_in[2];
  const float* Wout = (const float*)d_in[3];
  const float* bout = (const float*)d_in[4];
  float* out = (float*)d_out;

  __bf16* Xb    = (__bf16*)d_ws;
  __bf16* Wqkvb = Xb + (size_t)MROWS * DMODEL;
  __bf16* Woutb = Wqkvb + (size_t)3 * DMODEL * DMODEL;
  __bf16* QKVh  = Woutb + (size_t)DMODEL * DMODEL;
  __bf16* Attn  = QKVh + (size_t)3 * MROWS * DMODEL;

  {
    int n = MROWS * DMODEL;
    f32_to_bf16_kernel<<<n / 1024, 256, 0, stream>>>(x, Xb, n);
    n = 3 * DMODEL * DMODEL;
    f32_to_bf16_kernel<<<n / 1024, 256, 0, stream>>>(Wqkv, Wqkvb, n);
    n = DMODEL * DMODEL;
    f32_to_bf16_kernel<<<n / 1024, 256, 0, stream>>>(Wout, Woutb, n);
  }

  gemm_bf16_kernel<true><<<dim3(3 * DMODEL / 256, MROWS / 128), 256, 0,
                           stream>>>(Xb, Wqkvb, bqkv, (void*)QKVh, MROWS,
                                     3 * DMODEL, DMODEL);

  attn_kernel<<<dim3(S_LEN / 128, BATCH * NH), 256, 0, stream>>>(QKVh, Attn);

  gemm_bf16_kernel<false><<<dim3(DMODEL / 256, MROWS / 128), 256, 0, stream>>>(
      Attn, Woutb, bout, (void*)out, MROWS, DMODEL, DMODEL);
}
